// MultiModalPredictionHead_31585189495112
// MI455X (gfx1250) — compile-verified, matrix operand order checked by reading, not run on hardware
//
#include <hip/hip_runtime.h>
#include <cstdint>
#include <cstddef>

typedef __attribute__((ext_vector_type(16))) __bf16 v16bf;
typedef __attribute__((ext_vector_type(8)))  float  v8f;

constexpr int kH  = 512;
constexpr int kT  = 80;
constexpr int kM  = 6;
constexpr int kOD = 2;
constexpr int kB  = 2048;
constexpr int kBT = 32;       // batch rows per workgroup
constexpr int kThreads = 256; // 8 waves (wave32)

// ---------- helpers ----------
__device__ __forceinline__ __bf16 f2bf(float f) {
  union { float f; uint32_t u; } v; v.f = f;
  uint32_t r = v.u + 0x7FFFu + ((v.u >> 16) & 1u);  // RNE
  unsigned short h = (unsigned short)(r >> 16);
  __bf16 o; __builtin_memcpy(&o, &h, 2); return o;
}
__device__ __forceinline__ float bf2f(__bf16 b) {
  unsigned short h; __builtin_memcpy(&h, &b, 2);
  union { uint32_t u; float f; } v; v.u = ((uint32_t)h) << 16; return v.f;
}
__device__ __forceinline__ float sigf(float x) { return 1.f / (1.f + __expf(-x)); }
__device__ __forceinline__ v8f vzero8() { v8f z = {0.f,0.f,0.f,0.f,0.f,0.f,0.f,0.f}; return z; }

// CDNA5 16-bit A/B fragment (16x32 tile) from row-major memory, stride ld (elems).
// Lane m(0-15): VGPR0-3 = K 0..7, VGPR4-7 = K 16..23; lanes 16-31: K 8..15 / 24..31.
__device__ __forceinline__ v16bf load_frag(const __bf16* base, int ld) {
  const int lane = threadIdx.x & 31;
  const __bf16* p = base + (lane & 15) * ld + ((lane >> 4) << 3);
  union { v16bf v; uint4 u[2]; } f;
  f.u[0] = *reinterpret_cast<const uint4*>(p);
  f.u[1] = *reinterpret_cast<const uint4*>(p + 16);
  return f.v;
}

__device__ __forceinline__ v8f wmma_bf16(v16bf a, v16bf b, v8f c) {
  return __builtin_amdgcn_wmma_f32_16x16x32_bf16(false, a, false, b, (short)0, c, false, false);
}

// ---------- kernel: f32 -> bf16 conversion ----------
__global__ void k_f32_to_bf16(const float* __restrict__ src, __bf16* __restrict__ dst,
                              long long n) {
  long long i = (long long)blockIdx.x * blockDim.x + threadIdx.x;
  const long long stride = (long long)gridDim.x * blockDim.x;
  for (; i < n; i += stride) dst[i] = f2bf(src[i]);
}

// ---------- kernel: time embeddings te[T][H] ----------
__global__ void k_time_emb(const float* __restrict__ tpW1, const float* __restrict__ tpb1,
                           const float* __restrict__ tpW2, const float* __restrict__ tpb2,
                           float* __restrict__ te) {
  const int t = blockIdx.x;
  const int tid = threadIdx.x;
  __shared__ float hsh[256];
  const float tn = (float)t / (float)(kT - 1);
  hsh[tid] = fmaxf(tn * tpW1[tid] + tpb1[tid], 0.f);
  __syncthreads();
  for (int k = tid; k < kH; k += 256) {
    float s = tpb2[k];
    const float* w = tpW2 + (size_t)k * 256;
    for (int j = 0; j < 256; ++j) s += hsh[j] * w[j];
    te[(size_t)t * kH + k] = s;
  }
}

// ---------- kernel: confidences / logits ----------
__global__ void k_conf(const float* __restrict__ x, const float* __restrict__ W1,
                       const float* __restrict__ b1, const float* __restrict__ W2,
                       const float* __restrict__ b2, float* __restrict__ conf,
                       float* __restrict__ logits) {
  const int wave = threadIdx.x >> 5, lane = threadIdx.x & 31;
  const int row = blockIdx.x * 8 + wave;
  __shared__ float hsh[8][256];
  const float* xr = x + (size_t)row * kH;
  for (int j = lane; j < 256; j += 32) {
    float s = b1[j];
    const float* w = W1 + (size_t)j * kH;
    for (int k = 0; k < kH; ++k) s += xr[k] * w[k];
    hsh[wave][j] = fmaxf(s, 0.f);
  }
  __syncthreads();
  if (lane == 0) {
    float lg[kM], e[kM];
    float mx = -1e30f;
    for (int c = 0; c < kM; ++c) {
      float s = b2[c];
      const float* w = W2 + (size_t)c * 256;
      for (int j = 0; j < 256; ++j) s += hsh[wave][j] * w[j];
      lg[c] = s;
      logits[(size_t)row * kM + c] = s;
      mx = fmaxf(mx, s * (1.f / 1.2f));
    }
    float sum = 0.f;
    for (int c = 0; c < kM; ++c) { e[c] = __expf(lg[c] * (1.f / 1.2f) - mx); sum += e[c]; }
    for (int c = 0; c < kM; ++c) conf[(size_t)row * kM + c] = e[c] / sum;
  }
}

// ---------- kernel: mode features + LSTM init (WMMA) ----------
__global__ void __launch_bounds__(kThreads, 1) k_modefeat(
    const __bf16* __restrict__ xbf,
    const __bf16* __restrict__ mW1, const float* __restrict__ mb1,
    const __bf16* __restrict__ mW2, const float* __restrict__ mb2,
    const __bf16* __restrict__ mW3, const float* __restrict__ mb3,
    const __bf16* __restrict__ iW, const float* __restrict__ ib,
    __bf16* __restrict__ mf, __bf16* __restrict__ initH, float* __restrict__ initC) {
  const int m = blockIdx.x;
  const int b0v = blockIdx.y * kBT;
  const int tid = threadIdx.x, wave = tid >> 5, lane = tid & 31;
  extern __shared__ __align__(16) char smem[];
  __bf16* t1 = (__bf16*)smem;            // [kBT][2H]
  __bf16* t2 = t1 + kBT * 2 * kH;        // [kBT][H]

  const __bf16* W1 = mW1 + (size_t)m * 2 * kH * kH;
  const __bf16* Wi = iW  + (size_t)m * 2 * kH * kH;
  // phase 1a: t1 = relu(x @ mW1.T + mb1)   (32 x 1024)
  for (int q = 0; q < 16; ++q) {
    const int tIdx = wave * 16 + q, rt = tIdx >> 6, ct = tIdx & 63;
    v8f acc = vzero8();
    const __bf16* A  = xbf + (size_t)(b0v + rt * 16) * kH;
    const __bf16* Bw = W1 + (size_t)(ct * 16) * kH;
    for (int k0 = 0; k0 < kH; k0 += 32)
      acc = wmma_bf16(load_frag(A + k0, kH), load_frag(Bw + k0, kH), acc);
    const int col = ct * 16 + (lane & 15);
    const float bias = mb1[(size_t)m * 2 * kH + col];
#pragma unroll
    for (int r = 0; r < 8; ++r) {
      const int row = rt * 16 + r + ((lane >> 4) << 3);
      const float v = acc[r] + bias;
      t1[row * 2 * kH + col] = f2bf(v > 0.f ? v : 0.f);
    }
  }
  // phase 1b: init = x @ iW.T + ib  -> h part bf16, c part f32
  for (int q = 0; q < 16; ++q) {
    const int tIdx = wave * 16 + q, rt = tIdx >> 6, ct = tIdx & 63;
    v8f acc = vzero8();
    const __bf16* A  = xbf + (size_t)(b0v + rt * 16) * kH;
    const __bf16* Bw = Wi + (size_t)(ct * 16) * kH;
    for (int k0 = 0; k0 < kH; k0 += 32)
      acc = wmma_bf16(load_frag(A + k0, kH), load_frag(Bw + k0, kH), acc);
    const int col = ct * 16 + (lane & 15);
    const float bias = ib[(size_t)m * 2 * kH + col];
#pragma unroll
    for (int r = 0; r < 8; ++r) {
      const int row = rt * 16 + r + ((lane >> 4) << 3);
      const float v = acc[r] + bias;
      const size_t gi = ((size_t)m * kB + b0v + row) * kH;
      if (col < kH) initH[gi + col] = f2bf(v);
      else          initC[gi + (col - kH)] = v;
    }
  }
  __syncthreads();
  // phase 2: t2 = relu(t1 @ mW2.T + mb2), K = 2H
  const __bf16* W2 = mW2 + (size_t)m * kH * 2 * kH;
  for (int q = 0; q < 8; ++q) {
    const int tIdx = wave * 8 + q, rt = tIdx >> 5, ct = tIdx & 31;
    v8f acc = vzero8();
    const __bf16* A  = t1 + rt * 16 * 2 * kH;
    const __bf16* Bw = W2 + (size_t)(ct * 16) * 2 * kH;
    for (int k0 = 0; k0 < 2 * kH; k0 += 32)
      acc = wmma_bf16(load_frag(A + k0, 2 * kH), load_frag(Bw + k0, 2 * kH), acc);
    const int col = ct * 16 + (lane & 15);
    const float bias = mb2[(size_t)m * kH + col];
#pragma unroll
    for (int r = 0; r < 8; ++r) {
      const int row = rt * 16 + r + ((lane >> 4) << 3);
      const float v = acc[r] + bias;
      t2[row * kH + col] = f2bf(v > 0.f ? v : 0.f);
    }
  }
  __syncthreads();
  // phase 3: mf = t2 @ mW3.T + mb3
  const __bf16* W3 = mW3 + (size_t)m * kH * kH;
  for (int q = 0; q < 8; ++q) {
    const int tIdx = wave * 8 + q, rt = tIdx >> 5, ct = tIdx & 31;
    v8f acc = vzero8();
    const __bf16* A  = t2 + rt * 16 * kH;
    const __bf16* Bw = W3 + (size_t)(ct * 16) * kH;
    for (int k0 = 0; k0 < kH; k0 += 32)
      acc = wmma_bf16(load_frag(A + k0, kH), load_frag(Bw + k0, kH), acc);
    const int col = ct * 16 + (lane & 15);
    const float bias = mb3[(size_t)m * kH + col];
#pragma unroll
    for (int r = 0; r < 8; ++r) {
      const int row = rt * 16 + r + ((lane >> 4) << 3);
      mf[((size_t)m * kB + b0v + row) * kH + col] = f2bf(acc[r] + bias);
    }
  }
}

// ---------- LSTM cell (one step, one 32-row tile; state in LDS) ----------
__device__ __forceinline__ void cell_step(
    const __bf16* __restrict__ X, __bf16* Hbuf, __bf16* Htmp, float* C,
    const __bf16* __restrict__ Wi, const __bf16* __restrict__ Wh,
    const float* __restrict__ bi, const float* __restrict__ bh) {
  const int tid = threadIdx.x;
  const int wave = tid >> 5, lane = tid & 31;
  for (int q = 0; q < 8; ++q) {
    const int tIdx = wave * 8 + q;
    const int rt = tIdx >> 5, ct = tIdx & 31;
    v8f ai = vzero8(), af = vzero8(), ag = vzero8(), ao = vzero8();
    const __bf16* Xrow = X + rt * 16 * kH;
    const __bf16* Hrow = Hbuf + rt * 16 * kH;
    const __bf16* WiI = Wi + (size_t)(0 * kH + ct * 16) * kH;
    const __bf16* WhI = Wh + (size_t)(0 * kH + ct * 16) * kH;
    const __bf16* WiF = Wi + (size_t)(1 * kH + ct * 16) * kH;
    const __bf16* WhF = Wh + (size_t)(1 * kH + ct * 16) * kH;
    const __bf16* WiG = Wi + (size_t)(2 * kH + ct * 16) * kH;
    const __bf16* WhG = Wh + (size_t)(2 * kH + ct * 16) * kH;
    const __bf16* WiO = Wi + (size_t)(3 * kH + ct * 16) * kH;
    const __bf16* WhO = Wh + (size_t)(3 * kH + ct * 16) * kH;
    for (int k0 = 0; k0 < kH; k0 += 32) {
      const v16bf aX = load_frag(Xrow + k0, kH);
      const v16bf aH = load_frag(Hrow + k0, kH);
      ai = wmma_bf16(aX, load_frag(WiI + k0, kH), ai);
      ai = wmma_bf16(aH, load_frag(WhI + k0, kH), ai);
      af = wmma_bf16(aX, load_frag(WiF + k0, kH), af);
      af = wmma_bf16(aH, load_frag(WhF + k0, kH), af);
      ag = wmma_bf16(aX, load_frag(WiG + k0, kH), ag);
      ag = wmma_bf16(aH, load_frag(WhG + k0, kH), ag);
      ao = wmma_bf16(aX, load_frag(WiO + k0, kH), ao);
      ao = wmma_bf16(aH, load_frag(WhO + k0, kH), ao);
    }
    const int col = ct * 16 + (lane & 15);
    const float bI = bi[0 * kH + col] + bh[0 * kH + col];
    const float bF = bi[1 * kH + col] + bh[1 * kH + col];
    const float bG = bi[2 * kH + col] + bh[2 * kH + col];
    const float bO = bi[3 * kH + col] + bh[3 * kH + col];
#pragma unroll
    for (int r = 0; r < 8; ++r) {
      const int row = rt * 16 + r + ((lane >> 4) << 3);
      const float iv = sigf(ai[r] + bI);
      const float fv = sigf(af[r] + bF);
      const float gv = tanhf(ag[r] + bG);
      const float ov = sigf(ao[r] + bO);
      const float cn = fv * C[row * kH + col] + iv * gv;  // each c elem owned by one wave
      C[row * kH + col] = cn;
      Htmp[row * kH + col] = f2bf(ov * tanhf(cn));
    }
  }
  __syncthreads();  // all reads of old Hbuf done
  // vectorized LDS copy Htmp -> Hbuf (b128 moves)
  {
    uint4* d = reinterpret_cast<uint4*>(Hbuf);
    const uint4* s = reinterpret_cast<const uint4*>(Htmp);
    for (int i = tid; i < kBT * kH / 8; i += kThreads) d[i] = s[i];
  }
  __syncthreads();
}

// ---------- kernel: 80-step 2-layer LSTM rollout (WMMA, state-resident) ----------
__global__ void __launch_bounds__(kThreads, 1) k_rollout(
    const __bf16* __restrict__ mf, const __bf16* __restrict__ initH,
    const float* __restrict__ initC, const float* __restrict__ te,
    const __bf16* __restrict__ W0i, const __bf16* __restrict__ W0h,
    const __bf16* __restrict__ W1i, const __bf16* __restrict__ W1h,
    const float* __restrict__ b0i, const float* __restrict__ b0h,
    const float* __restrict__ b1i, const float* __restrict__ b1h,
    const __bf16* __restrict__ oW1, const float* __restrict__ ob1,
    const float* __restrict__ oW2, const float* __restrict__ ob2,
    const float* __restrict__ fbW, const float* __restrict__ fbb,
    float* __restrict__ traj) {
  const int m = blockIdx.x;
  const int b0v = blockIdx.y * kBT;
  const int tid = threadIdx.x, wave = tid >> 5, lane = tid & 31;
  extern __shared__ __align__(16) char smem[];
  __bf16* xin = (__bf16*)smem;               // [kBT][H] bf16
  __bf16* h0  = xin + kBT * kH;
  __bf16* h1  = h0 + kBT * kH;
  __bf16* hid = h1 + kBT * kH;               // doubles as H-staging
  __bf16* mfs = hid + kBT * kH;              // mode features, pinned for all 80 steps
  float*  c0  = (float*)(mfs + kBT * kH);    // [kBT][H] f32
  float*  c1  = c0 + kBT * kH;
  float*  prev = c1 + kBT * kH;              // [kBT][OD]
  float*  part = prev + kBT * kOD;           // [kBT*OD][4] reduction scratch

  for (int i = tid; i < kBT * kH; i += kThreads) {
    const int r = i / kH, k = i - r * kH;
    const size_t gi = ((size_t)m * kB + b0v + r) * kH + k;
    const __bf16 hv = initH[gi];
    h0[i] = hv; h1[i] = hv;
    mfs[i] = mf[gi];
    const float cv = initC[gi];
    c0[i] = cv; c1[i] = cv;
  }
  if (tid < kBT * kOD) prev[tid] = 0.f;
  __syncthreads();

  const __bf16* Wi0 = W0i + (size_t)m * 4 * kH * kH;
  const __bf16* Wh0 = W0h + (size_t)m * 4 * kH * kH;
  const __bf16* Wi1 = W1i + (size_t)m * 4 * kH * kH;
  const __bf16* Wh1 = W1h + (size_t)m * 4 * kH * kH;
  const float* bi0 = b0i + (size_t)m * 4 * kH;
  const float* bh0 = b0h + (size_t)m * 4 * kH;
  const float* bi1 = b1i + (size_t)m * 4 * kH;
  const float* bh1 = b1h + (size_t)m * 4 * kH;
  const __bf16* Wp  = oW1 + (size_t)m * kH * kH;
  const float* ob1m = ob1 + (size_t)m * kH;
  const float* fbWm = fbW + (size_t)m * (kH / 2) * kOD;
  const float* fbbm = fbb + (size_t)m * (kH / 2);

  for (int t = 0; t < kT; ++t) {
    // prefetch next step's time-embedding row (global_prefetch_b8)
    if (t + 1 < kT && tid < 32)
      __builtin_prefetch(te + (size_t)(t + 1) * kH + tid * 16, 0, 1);

    // xin = mf + 0.5*te[t] + 0.3*pad(prev @ fbW.T + fbb)   (mf from LDS)
    for (int i = tid; i < kBT * kH; i += kThreads) {
      const int r = i / kH, k = i - r * kH;
      float v = bf2f(mfs[i]) + 0.5f * te[(size_t)t * kH + k];
      if (k < kH / 2) {
        const float fbv = fbWm[k * kOD + 0] * prev[r * kOD + 0] +
                          fbWm[k * kOD + 1] * prev[r * kOD + 1] + fbbm[k];
        v += 0.3f * fbv;
      }
      xin[i] = f2bf(v);
    }
    __syncthreads();

    cell_step(xin, h0, hid, c0, Wi0, Wh0, bi0, bh0);
    cell_step(h0,  h1, hid, c1, Wi1, Wh1, bi1, bh1);

    // hidden = relu(h1 @ oW1.T + ob1)  -> hid
    for (int q = 0; q < 8; ++q) {
      const int tIdx = wave * 8 + q, rt = tIdx >> 5, ct = tIdx & 31;
      v8f acc = vzero8();
      for (int k0 = 0; k0 < kH; k0 += 32)
        acc = wmma_bf16(load_frag(h1 + rt * 16 * kH + k0, kH),
                        load_frag(Wp + (size_t)(ct * 16) * kH + k0, kH), acc);
      const int col = ct * 16 + (lane & 15);
      const float bias = ob1m[col];
#pragma unroll
      for (int r = 0; r < 8; ++r) {
        const int row = rt * 16 + r + ((lane >> 4) << 3);
        const float v = acc[r] + bias;
        hid[row * kH + col] = f2bf(v > 0.f ? v : 0.f);
      }
    }
    __syncthreads();

    // out = hidden @ oW2.T + ob2  (OD = 2): 256 threads, 4 partials per (row,od)
    {
      const int p = tid >> 2;          // 0..63 -> (row, od)
      const int sub = tid & 3;         // 128-elem K slice
      const int r = p >> 1, od = p & 1;
      const float* w2 = oW2 + ((size_t)m * kOD + od) * kH + sub * 128;
      const __bf16* hrow = hid + r * kH + sub * 128;
      float s = 0.f;
      for (int k = 0; k < 128; ++k) s += bf2f(hrow[k]) * w2[k];
      part[p * 4 + sub] = s;
    }
    __syncthreads();
    if (tid < kBT * kOD) {
      const int r = tid >> 1, od = tid & 1;
      const float s = ob2[(size_t)m * kOD + od] + part[tid * 4 + 0] + part[tid * 4 + 1] +
                      part[tid * 4 + 2] + part[tid * 4 + 3];
      prev[tid] = s;
      const size_t off = (((size_t)(b0v + r) * kM + m) * kT + t) * kOD + od;
      traj[off] = (t == 0) ? 0.f : s;  // traj[:,0,:] is zeroed; carry still uses s
    }
    __syncthreads();
  }
}

// ---------- host ----------
extern "C" void kernel_launch(void* const* d_in, const int* in_sizes, int n_in,
                              void* d_out, int out_size, void* d_ws, size_t ws_size,
                              hipStream_t stream) {
  (void)in_sizes; (void)n_in; (void)out_size; (void)ws_size;
  const float* x    = (const float*)d_in[0];
  const float* mdW1 = (const float*)d_in[1];
  const float* mdb1 = (const float*)d_in[2];
  const float* mdW2 = (const float*)d_in[3];
  const float* mdb2 = (const float*)d_in[4];
  const float* mdW3 = (const float*)d_in[5];
  const float* mdb3 = (const float*)d_in[6];
  const float* ihW  = (const float*)d_in[7];
  const float* ihb  = (const float*)d_in[8];
  const float* tpW1 = (const float*)d_in[9];
  const float* tpb1 = (const float*)d_in[10];
  const float* tpW2 = (const float*)d_in[11];
  const float* tpb2 = (const float*)d_in[12];
  const float* fbW  = (const float*)d_in[13];
  const float* fbb  = (const float*)d_in[14];
  const float* W0i  = (const float*)d_in[15];
  const float* b0i  = (const float*)d_in[16];
  const float* W0h  = (const float*)d_in[17];
  const float* b0h  = (const float*)d_in[18];
  const float* W1i  = (const float*)d_in[19];
  const float* b1i  = (const float*)d_in[20];
  const float* W1h  = (const float*)d_in[21];
  const float* b1h  = (const float*)d_in[22];
  const float* oW1  = (const float*)d_in[23];
  const float* ob1  = (const float*)d_in[24];
  const float* oW2  = (const float*)d_in[25];
  const float* ob2  = (const float*)d_in[26];
  const float* cfW1 = (const float*)d_in[27];
  const float* cfb1 = (const float*)d_in[28];
  const float* cfW2 = (const float*)d_in[29];
  const float* cfb2 = (const float*)d_in[30];

  char* w = (char*)d_ws;
  size_t off = 0;
  auto take = [&](size_t bytes) -> void* {
    void* p = w + off;
    off += (bytes + 255) & ~(size_t)255;
    return p;
  };
  __bf16* xbf    = (__bf16*)take((size_t)kB * kH * 2);
  __bf16* mdW1b  = (__bf16*)take((size_t)kM * 2 * kH * kH * 2);
  __bf16* mdW2b  = (__bf16*)take((size_t)kM * kH * 2 * kH * 2);
  __bf16* mdW3b  = (__bf16*)take((size_t)kM * kH * kH * 2);
  __bf16* ihWb   = (__bf16*)take((size_t)kM * 2 * kH * kH * 2);
  __bf16* W0ib   = (__bf16*)take((size_t)kM * 4 * kH * kH * 2);
  __bf16* W0hb   = (__bf16*)take((size_t)kM * 4 * kH * kH * 2);
  __bf16* W1ib   = (__bf16*)take((size_t)kM * 4 * kH * kH * 2);
  __bf16* W1hb   = (__bf16*)take((size_t)kM * 4 * kH * kH * 2);
  __bf16* oW1b   = (__bf16*)take((size_t)kM * kH * kH * 2);
  __bf16* mfb    = (__bf16*)take((size_t)kM * kB * kH * 2);
  __bf16* initHb = (__bf16*)take((size_t)kM * kB * kH * 2);
  float*  teb    = (float*)take((size_t)kT * kH * 4);
  float*  initCb = (float*)take((size_t)kM * kB * kH * 4);

  auto conv = [&](const float* s, __bf16* dst, long long n) {
    int blocks = (int)((n + 2047) / 2048);
    if (blocks > 4096) blocks = 4096;
    k_f32_to_bf16<<<blocks, 256, 0, stream>>>(s, dst, n);
  };
  conv(x, xbf, (long long)kB * kH);
  conv(mdW1, mdW1b, (long long)kM * 2 * kH * kH);
  conv(mdW2, mdW2b, (long long)kM * kH * 2 * kH);
  conv(mdW3, mdW3b, (long long)kM * kH * kH);
  conv(ihW,  ihWb,  (long long)kM * 2 * kH * kH);
  conv(W0i,  W0ib,  (long long)kM * 4 * kH * kH);
  conv(W0h,  W0hb,  (long long)kM * 4 * kH * kH);
  conv(W1i,  W1ib,  (long long)kM * 4 * kH * kH);
  conv(W1h,  W1hb,  (long long)kM * 4 * kH * kH);
  conv(oW1,  oW1b,  (long long)kM * kH * kH);

  k_time_emb<<<kT, 256, 0, stream>>>(tpW1, tpb1, tpW2, tpb2, teb);

  const size_t mfeat_smem = (size_t)kBT * 2 * kH * 2 + (size_t)kBT * kH * 2;  // 96 KB
  (void)hipFuncSetAttribute(reinterpret_cast<const void*>(k_modefeat),
                            hipFuncAttributeMaxDynamicSharedMemorySize, (int)mfeat_smem);
  k_modefeat<<<dim3(kM, kB / kBT), kThreads, mfeat_smem, stream>>>(
      xbf, mdW1b, mdb1, mdW2b, mdb2, mdW3b, mdb3, ihWb, ihb, mfb, initHb, initCb);

  float* traj   = (float*)d_out;
  float* conf   = traj + (size_t)kB * kM * kT * kOD;
  float* logits = conf + (size_t)kB * kM;
  k_conf<<<kB / 8, 256, 0, stream>>>(x, cfW1, cfb1, cfW2, cfb2, conf, logits);

  // LDS: xin/h0/h1/hid/mf (bf16) + c0/c1 (f32) + prev + partials  ~= 296 KB <= 320 KB
  const size_t roll_smem = (size_t)5 * kBT * kH * 2 + (size_t)2 * kBT * kH * 4 +
                           (size_t)kBT * kOD * 4 + (size_t)kBT * kOD * 4 * 4;
  (void)hipFuncSetAttribute(reinterpret_cast<const void*>(k_rollout),
                            hipFuncAttributeMaxDynamicSharedMemorySize, (int)roll_smem);
  k_rollout<<<dim3(kM, kB / kBT), kThreads, roll_smem, stream>>>(
      mfb, initHb, initCb, teb, W0ib, W0hb, W1ib, W1hb, b0i, b0h, b1i, b1h,
      oW1b, ob1, oW2, ob2, fbW, fbb, traj);
}
